// Attention_34359739056
// MI455X (gfx1250) — matrix operand order repaired and checked by reading, not compiled, not run on hardware
//
#include <hip/hip_runtime.h>

// ---------------------------------------------------------------------------
// ViT multi-head attention block on gfx1250 (MI455X), all GEMMs via
// v_wmma_f32_16x16x32_bf16 (wave32 WMMA), flash-style two-pass softmax.
// ---------------------------------------------------------------------------

typedef unsigned short ushort_t;
typedef __attribute__((ext_vector_type(16))) __bf16 v16bf;
typedef __attribute__((ext_vector_type(8)))  float  v8f;

#define DIM     768
#define NH      12
#define HD      64
#define SEQ     1024
#define BATCH   8
#define QKV_N   (3 * DIM)        // 2304
#define TOKENS  (BATCH * SEQ)    // 8192
#define ATT_SCALE 0.125f         // 64^-0.5

__device__ __forceinline__ ushort_t f2bf(float f) {
  unsigned u = __float_as_uint(f);
  u += 0x7FFFu + ((u >> 16) & 1u);     // round-to-nearest-even
  return (ushort_t)(u >> 16);
}

union FragU { v16bf v; uint4 q[2]; };

// A-fragment (16x32 bf16): lane holds row M=lane&15; half h=lane>>4 holds
// K in {h*8 .. h*8+7} and {16+h*8 .. 16+h*8+7}.  p -> &M[row0][k0], row-major.
__device__ __forceinline__ v16bf load_a_frag(const ushort_t* p, int ld, int lane) {
  int r = lane & 15, hh = (lane >> 4) & 1;
  const ushort_t* q = p + (size_t)r * ld + hh * 8;
  FragU u;
  u.q[0] = *(const uint4*)(q);        // K = h*8 + 0..7
  u.q[1] = *(const uint4*)(q + 16);   // K = 16 + h*8 + 0..7
  return u.v;
}

// B-fragment (32x16 bf16) loaded from B^T stored row-major [N][K]:
// lane holds column N=lane&15; half h holds K in {h*8 .. h*8+7} and {16+h*8 .. 16+h*8+7}.
__device__ __forceinline__ v16bf load_b_frag(const ushort_t* p, int ld, int lane) {
  int r = lane & 15, hh = (lane >> 4) & 1;
  const ushort_t* q = p + (size_t)r * ld + hh * 8;
  FragU u;
  u.q[0] = *(const uint4*)(q);        // K = h*8 + 0..7
  u.q[1] = *(const uint4*)(q + 16);   // K = 16 + h*8 + 0..7
  return u.v;
}

__device__ __forceinline__ v8f wmma_bf16(v16bf a, v16bf b, v8f c) {
  return __builtin_amdgcn_wmma_f32_16x16x32_bf16(
      false, a, false, b, (short)0, c, false, false);
}

// ------------------------------- converters --------------------------------

__global__ void cvt_f32_bf16_kernel(const float* __restrict__ src,
                                    ushort_t* __restrict__ dst, int n) {
  int i = blockIdx.x * blockDim.x + threadIdx.x;
  if (i < n) dst[i] = f2bf(src[i]);
}

// src [R][C] fp32 row-major  ->  dst [C][R] bf16 row-major (i.e. transposed)
__global__ void cvt_t_bf16_kernel(const float* __restrict__ src,
                                  ushort_t* __restrict__ dst, int R, int C) {
  int i = blockIdx.x * blockDim.x + threadIdx.x;
  if (i < R * C) {
    int r = i / C, c = i - r * C;
    dst[(size_t)c * R + r] = f2bf(src[i]);
  }
}

// ------------------------------- QKV GEMM ----------------------------------
// qkv = x(bf16)[8192x768] @ w_qkv[768x2304]; scatter into Q/K head-major and
// V transposed per head.  One wave computes a 16x64 output tile.

__global__ __launch_bounds__(256) void qkv_gemm_kernel(
    const ushort_t* __restrict__ xb, const ushort_t* __restrict__ wT,
    ushort_t* __restrict__ Qd, ushort_t* __restrict__ Kd,
    ushort_t* __restrict__ Vt) {
  int wave = blockIdx.x * 8 + (threadIdx.x >> 5);
  int lane = threadIdx.x & 31;
  const int numN4 = QKV_N / 64;                 // 36
  int mt = wave / numN4;
  int n4 = wave - mt * numN4;
  if (mt >= TOKENS / 16) return;

  const ushort_t* A  = xb + (size_t)mt * 16 * DIM;
  const ushort_t* BT = wT + (size_t)n4 * 64 * DIM;

  v8f acc0 = {}, acc1 = {}, acc2 = {}, acc3 = {};
  for (int k = 0; k < DIM; k += 32) {
    v16bf a  = load_a_frag(A + k, DIM, lane);
    v16bf b0 = load_b_frag(BT + (size_t)0 * 16 * DIM + k, DIM, lane);
    v16bf b1 = load_b_frag(BT + (size_t)1 * 16 * DIM + k, DIM, lane);
    v16bf b2 = load_b_frag(BT + (size_t)2 * 16 * DIM + k, DIM, lane);
    v16bf b3 = load_b_frag(BT + (size_t)3 * 16 * DIM + k, DIM, lane);
    acc0 = wmma_bf16(a, b0, acc0);
    acc1 = wmma_bf16(a, b1, acc1);
    acc2 = wmma_bf16(a, b2, acc2);
    acc3 = wmma_bf16(a, b3, acc3);
  }

  int hh = (lane >> 4) & 1;
  int cl = lane & 15;
  v8f accs[4] = {acc0, acc1, acc2, acc3};
#pragma unroll
  for (int d = 0; d < 4; ++d) {
    int n = n4 * 64 + d * 16 + cl;      // column in [0, 2304)
    int s = n / DIM;                    // 0=q, 1=k, 2=v
    int rem = n - s * DIM;
    int head = rem >> 6;
    int dd = rem & 63;
#pragma unroll
    for (int j = 0; j < 8; ++j) {
      int r = mt * 16 + j + 8 * hh;     // token row in [0, 8192)
      int b = r >> 10, t = r & 1023;
      ushort_t val = f2bf(accs[d][j]);
      size_t bh = (size_t)(b * NH + head);
      if (s == 0)      Qd[(bh * SEQ + t) * HD + dd] = val;
      else if (s == 1) Kd[(bh * SEQ + t) * HD + dd] = val;
      else             Vt[(bh * HD + dd) * SEQ + t] = val;   // V transposed
    }
  }
}

// ------------------------------- attention ---------------------------------
// One wave handles one (b, h, 16-query tile).  Pass 1: row max of scaled
// logits.  Pass 2: exp, row-sum, and P·V via WMMA (P re-layouted C->A frag
// through LDS).  4 waves / 128-thread block, 4 KB static LDS.

__global__ __launch_bounds__(128) void attn_kernel(
    const ushort_t* __restrict__ Qd, const ushort_t* __restrict__ Kd,
    const ushort_t* __restrict__ Vt, ushort_t* __restrict__ A2) {
  __shared__ __align__(16) ushort_t Plds[4][16 * 32];

  int wid  = threadIdx.x >> 5;
  int lane = threadIdx.x & 31;
  int gw = blockIdx.x * 4 + wid;
  int qt = gw & 63;
  int h  = (gw >> 6) % NH;
  int b  = gw / (64 * NH);
  size_t bh = (size_t)(b * NH + h);

  const ushort_t* Qb = Qd + (bh * SEQ + (size_t)qt * 16) * HD;
  const ushort_t* Kb = Kd + bh * SEQ * HD;
  const ushort_t* Vb = Vt + bh * HD * SEQ;

  v16bf q0 = load_a_frag(Qb + 0,  HD, lane);
  v16bf q1 = load_a_frag(Qb + 32, HD, lane);

  // ---- pass 1: row max of scaled logits ----
  float m[8];
#pragma unroll
  for (int j = 0; j < 8; ++j) m[j] = -3.0e38f;

  for (int kt = 0; kt < SEQ / 16; ++kt) {
    const ushort_t* Kp = Kb + (size_t)kt * 16 * HD;
    v16bf k0 = load_b_frag(Kp + 0,  HD, lane);
    v16bf k1 = load_b_frag(Kp + 32, HD, lane);
    v8f c = {};
    c = wmma_bf16(q0, k0, c);
    c = wmma_bf16(q1, k1, c);
#pragma unroll
    for (int j = 0; j < 8; ++j) m[j] = fmaxf(m[j], c[j] * ATT_SCALE);
  }
#pragma unroll
  for (int j = 0; j < 8; ++j) {      // reduce across the 16-lane column group
    float v = m[j];
    v = fmaxf(v, __shfl_xor(v, 1, 32));
    v = fmaxf(v, __shfl_xor(v, 2, 32));
    v = fmaxf(v, __shfl_xor(v, 4, 32));
    v = fmaxf(v, __shfl_xor(v, 8, 32));
    m[j] = v;
  }

  // ---- pass 2: exp, row sums, P·V ----
  float rs[8];
#pragma unroll
  for (int j = 0; j < 8; ++j) rs[j] = 0.0f;
  v8f o0 = {}, o1 = {}, o2 = {}, o3 = {};

  int hh = (lane >> 4) & 1;
  int cl = lane & 15;
  ushort_t* Pw = &Plds[wid][0];

  for (int ks = 0; ks < SEQ / 32; ++ks) {
#pragma unroll
    for (int sub = 0; sub < 2; ++sub) {
      int kt = ks * 2 + sub;
      const ushort_t* Kp = Kb + (size_t)kt * 16 * HD;
      v16bf k0 = load_b_frag(Kp + 0,  HD, lane);
      v16bf k1 = load_b_frag(Kp + 32, HD, lane);
      v8f c = {};
      c = wmma_bf16(q0, k0, c);
      c = wmma_bf16(q1, k1, c);
#pragma unroll
      for (int j = 0; j < 8; ++j) {
        float p = __expf(c[j] * ATT_SCALE - m[j]);
        rs[j] += p;
        Pw[(j + 8 * hh) * 32 + sub * 16 + cl] = f2bf(p);   // C-frag -> LDS
      }
    }
    __syncthreads();                       // make P visible across lanes
    v16bf pa = load_a_frag(Pw, 32, lane);  // LDS -> A-frag (16x32 keys)
    const ushort_t* Vp = Vb + ks * 32;
    v16bf v0 = load_b_frag(Vp + (size_t)0 * 16 * SEQ, SEQ, lane);
    v16bf v1 = load_b_frag(Vp + (size_t)1 * 16 * SEQ, SEQ, lane);
    v16bf v2 = load_b_frag(Vp + (size_t)2 * 16 * SEQ, SEQ, lane);
    v16bf v3 = load_b_frag(Vp + (size_t)3 * 16 * SEQ, SEQ, lane);
    o0 = wmma_bf16(pa, v0, o0);
    o1 = wmma_bf16(pa, v1, o1);
    o2 = wmma_bf16(pa, v2, o2);
    o3 = wmma_bf16(pa, v3, o3);
    __syncthreads();                       // reads done before next overwrite
  }

#pragma unroll
  for (int j = 0; j < 8; ++j) {            // row-sum reduction, then 1/sum
    float v = rs[j];
    v += __shfl_xor(v, 1, 32);
    v += __shfl_xor(v, 2, 32);
    v += __shfl_xor(v, 4, 32);
    v += __shfl_xor(v, 8, 32);
    rs[j] = 1.0f / v;
  }

  v8f oo[4] = {o0, o1, o2, o3};
#pragma unroll
  for (int d = 0; d < 4; ++d) {
#pragma unroll
    for (int j = 0; j < 8; ++j) {
      int t  = qt * 16 + j + 8 * hh;
      int ch = h * HD + d * 16 + cl;
      A2[((size_t)b * SEQ + t) * DIM + ch] = f2bf(oo[d][j] * rs[j]);
    }
  }
}

// ------------------------------- proj GEMM ---------------------------------
// out = attn_out(bf16)[8192x768] @ w_proj[768x768] + b_proj, fp32 output.

__global__ __launch_bounds__(256) void proj_gemm_kernel(
    const ushort_t* __restrict__ A2, const ushort_t* __restrict__ wT,
    const float* __restrict__ bias, float* __restrict__ out) {
  int wave = blockIdx.x * 8 + (threadIdx.x >> 5);
  int lane = threadIdx.x & 31;
  const int numN4 = DIM / 64;                  // 12
  int mt = wave / numN4;
  int n4 = wave - mt * numN4;
  if (mt >= TOKENS / 16) return;

  const ushort_t* A  = A2 + (size_t)mt * 16 * DIM;
  const ushort_t* BT = wT + (size_t)n4 * 64 * DIM;

  v8f acc0 = {}, acc1 = {}, acc2 = {}, acc3 = {};
  for (int k = 0; k < DIM; k += 32) {
    v16bf a  = load_a_frag(A + k, DIM, lane);
    v16bf b0 = load_b_frag(BT + (size_t)0 * 16 * DIM + k, DIM, lane);
    v16bf b1 = load_b_frag(BT + (size_t)1 * 16 * DIM + k, DIM, lane);
    v16bf b2 = load_b_frag(BT + (size_t)2 * 16 * DIM + k, DIM, lane);
    v16bf b3 = load_b_frag(BT + (size_t)3 * 16 * DIM + k, DIM, lane);
    acc0 = wmma_bf16(a, b0, acc0);
    acc1 = wmma_bf16(a, b1, acc1);
    acc2 = wmma_bf16(a, b2, acc2);
    acc3 = wmma_bf16(a, b3, acc3);
  }

  int hh = (lane >> 4) & 1;
  int cl = lane & 15;
  v8f accs[4] = {acc0, acc1, acc2, acc3};
#pragma unroll
  for (int d = 0; d < 4; ++d) {
    int n = n4 * 64 + d * 16 + cl;
    float bv = bias[n];
#pragma unroll
    for (int j = 0; j < 8; ++j) {
      int r = mt * 16 + j + 8 * hh;
      out[(size_t)r * DIM + n] = accs[d][j] + bv;
    }
  }
}

// ------------------------------- launcher ----------------------------------

extern "C" void kernel_launch(void* const* d_in, const int* in_sizes, int n_in,
                              void* d_out, int out_size, void* d_ws, size_t ws_size,
                              hipStream_t stream) {
  (void)in_sizes; (void)n_in; (void)out_size; (void)ws_size;

  const float* x      = (const float*)d_in[0];   // [8,1024,768]
  const float* w_qkv  = (const float*)d_in[1];   // [768,2304]
  const float* w_proj = (const float*)d_in[2];   // [768,768]
  const float* b_proj = (const float*)d_in[3];   // [768]
  float* out = (float*)d_out;                    // [8,1024,768]

  char* ws = (char*)d_ws;
  size_t off = 0;
  ushort_t* xb     = (ushort_t*)(ws + off); off += (size_t)TOKENS * DIM * 2;  // 12.6 MB
  ushort_t* wqkvT  = (ushort_t*)(ws + off); off += (size_t)QKV_N * DIM * 2;   //  3.5 MB
  ushort_t* wprojT = (ushort_t*)(ws + off); off += (size_t)DIM * DIM * 2;     //  1.2 MB
  ushort_t* Qd     = (ushort_t*)(ws + off); off += (size_t)TOKENS * DIM * 2;  // [B,H,N,Dh]
  ushort_t* Kd     = (ushort_t*)(ws + off); off += (size_t)TOKENS * DIM * 2;  // [B,H,N,Dh]
  ushort_t* Vt     = (ushort_t*)(ws + off); off += (size_t)TOKENS * DIM * 2;  // [B,H,Dh,N]
  ushort_t* A2     = (ushort_t*)(ws + off); off += (size_t)TOKENS * DIM * 2;  // [B,N,C]

  // 1) fp32 -> bf16 conversions (+ weight transposes for B-fragment loads)
  cvt_f32_bf16_kernel<<<(TOKENS * DIM + 255) / 256, 256, 0, stream>>>(x, xb, TOKENS * DIM);
  cvt_t_bf16_kernel<<<(DIM * QKV_N + 255) / 256, 256, 0, stream>>>(w_qkv, wqkvT, DIM, QKV_N);
  cvt_t_bf16_kernel<<<(DIM * DIM + 255) / 256, 256, 0, stream>>>(w_proj, wprojT, DIM, DIM);

  // 2) QKV projection: 512 M-tiles x 36 N4-tiles = 18432 waves / 8 per block
  qkv_gemm_kernel<<<(TOKENS / 16) * (QKV_N / 64) / 8, 256, 0, stream>>>(xb, wqkvT, Qd, Kd, Vt);

  // 3) attention: 8*12*64 = 6144 waves, 4 per block
  attn_kernel<<<BATCH * NH * (SEQ / 16) / 4, 128, 0, stream>>>(Qd, Kd, Vt, A2);

  // 4) output projection + bias: 512 x 12 = 6144 waves / 8 per block
  proj_gemm_kernel<<<(TOKENS / 16) * (DIM / 64) / 8, 256, 0, stream>>>(A2, wprojT, b_proj, out);
}
